// EncoderModel_21337397527113
// MI455X (gfx1250) — matrix operand order repaired and checked by reading, not compiled, not run on hardware
//
#include <hip/hip_runtime.h>

typedef __attribute__((ext_vector_type(16))) _Float16 v16h;
typedef __attribute__((ext_vector_type(8)))  _Float16 h8;
typedef __attribute__((ext_vector_type(8)))  float    v8f;

namespace {

constexpr int NB   = 128;       // batch
constexpr int SSZ  = 19;        // spatial
constexpr int PP   = SSZ * SSZ; // 361
constexpr int MPIX = NB * PP;   // 46208 conv GEMM rows

__device__ __forceinline__ float fast_act(float v, int act) {
  if (act == 1) return fmaxf(v, 0.0f);
  if (act == 2) {                       // tanh(x) = 1 - 2/(exp(2x)+1)
    float e = __expf(2.0f * v);
    return 1.0f - 2.0f / (e + 1.0f);
  }
  return v;
}

// ---------------------------------------------------------------------------
// WMMA GEMM: C[M,N] = act( A' * Bk + bias )
//   mode 0: A is plain [M,K] row-major f32
//   mode 1: implicit im2col from activations [NB, PP, Cin] (NPC)
//   mode 2: implicit im2col from NCHW input [NB, Cin, 19, 19]
//   (k = tap*Cin + c, tap = 3*(dy+1)+(dx+1))
// Block = 256 threads = 8 waves; block tile 64(M) x 64(N); each wave owns a
// 16x32 strip (A fragment reused across two WMMAs). K loop in steps of 32 with
// f32->f16 LDS staging; all row decode hoisted, tap/c updated incrementally.
// ---------------------------------------------------------------------------
__global__ __launch_bounds__(256) void gemm_wmma_kernel(
    const float* __restrict__ A, const float* __restrict__ Bk,
    const float* __restrict__ bias, float* __restrict__ C,
    int M, int N, int K, int mode, int Cin, int act)
{
  __shared__ alignas(16) _Float16 As [64][40];   // [m][k]
  __shared__ alignas(16) _Float16 BsT[64][40];   // [n][k] (transposed)

  const int tid  = threadIdx.x;
  const int lane = tid & 31;
  const int wave = tid >> 5;
  const int mblk = blockIdx.y * 64;
  const int nblk = blockIdx.x * 64;
  const int mloc = (wave & 3) * 16;   // wave row base in tile
  const int nloc = (wave >> 2) * 32;  // wave col base in tile (32 wide)

  // ---- hoisted A-staging metadata: kk fixed = tid&31, rows fixed ----
  const int kkA   = tid & 31;
  const int rowA0 = tid >> 5;         // rows rowA0 + 8j, j = 0..7
  int  bA[8], yA[8], xA[8];
  const float* aRow[8];
  #pragma unroll
  for (int j = 0; j < 8; ++j) {
    int r  = mblk + rowA0 + 8 * j;
    int rr = (r < M) ? r : 0;
    int b  = rr / PP, p = rr - b * PP;
    int y  = p / SSZ;
    bA[j] = b; yA[j] = y; xA[j] = p - y * SSZ;
    aRow[j] = A + (long)rr * K + kkA;           // mode 0 base
  }
  // tap/c for k = kkA, stepped by +32 per K iteration (no divisions in loop)
  int tap = 0, cc = kkA;
  if (mode != 0) { tap = kkA / Cin; cc = kkA - tap * Cin; }

  // ---- hoisted B-staging metadata: n fixed = tid&63, k rows fixed ----
  const int nB  = tid & 63;
  const int kB0 = tid >> 6;           // k rows kB0 + 4j, j = 0..7
  const int gnB = nblk + nB;

  v8f acc0 = {}, acc1 = {};

  #pragma unroll 2
  for (int k0 = 0; k0 < K; k0 += 32) {
    // ---- stage A tile (64m x 32k) ----
    const bool kOkA = (k0 + kkA) < K;
    int dy = 0, dx = 0;
    if (mode != 0) { dy = tap / 3 - 1; dx = tap - (tap / 3) * 3 - 1; }
    #pragma unroll
    for (int j = 0; j < 8; ++j) {
      float v = 0.0f;
      int r = mblk + rowA0 + 8 * j;
      if (r < M && kOkA) {
        if (mode == 0) {
          v = aRow[j][k0];
        } else {
          int yy = yA[j] + dy, xx = xA[j] + dx;
          if (yy >= 0 && yy < SSZ && xx >= 0 && xx < SSZ) {
            v = (mode == 1) ? A[((long)(bA[j] * PP + yy * SSZ + xx)) * Cin + cc]
                            : A[((long)(bA[j] * Cin + cc)) * PP + yy * SSZ + xx];
          }
        }
      }
      As[rowA0 + 8 * j][kkA] = (_Float16)v;
    }
    if (mode != 0) {                    // incremental k += 32 -> tap/c update
      cc += 32;
      while (cc >= Cin) { cc -= Cin; ++tap; }
    }
    // ---- stage B tile (32k x 64n) into transposed LDS ----
    #pragma unroll
    for (int j = 0; j < 8; ++j) {
      int kk = kB0 + 4 * j;
      int k  = k0 + kk;
      float v = (k < K && gnB < N) ? Bk[(long)k * N + gnB] : 0.0f;
      BsT[nB][kk] = (_Float16)v;
    }
    if (k0 + 32 < K && gnB < N)
      __builtin_prefetch(&Bk[(long)(k0 + 32) * N + gnB], 0, 0);
    __syncthreads();

    // ---- fragments via aligned 16B LDS vector loads ----
    // A 16x32 f16: lanes 0-15 hold K 0..15, lanes 16-31 K 16..31
    const int hlane = lane & 15;
    const int hsel  = lane >> 4;
    const h8* ap = (const h8*)&As[mloc + hlane][hsel * 16];
    v16h af = __builtin_shufflevector(ap[0], ap[1],
        0,1,2,3,4,5,6,7,8,9,10,11,12,13,14,15);
    // B 32x16 f16: lanes 0-15 N=lane K 0..15, lanes 16-31 N=lane-16 K 16..31
    const h8* bp0 = (const h8*)&BsT[nloc + hlane][hsel * 16];
    const h8* bp1 = (const h8*)&BsT[nloc + 16 + hlane][hsel * 16];
    v16h bf0 = __builtin_shufflevector(bp0[0], bp0[1],
        0,1,2,3,4,5,6,7,8,9,10,11,12,13,14,15);
    v16h bf1 = __builtin_shufflevector(bp1[0], bp1[1],
        0,1,2,3,4,5,6,7,8,9,10,11,12,13,14,15);

    acc0 = __builtin_amdgcn_wmma_f32_16x16x32_f16(false, af, false, bf0,
                                                  (short)0, acc0, false, false);
    acc1 = __builtin_amdgcn_wmma_f32_16x16x32_f16(false, af, false, bf1,
                                                  (short)0, acc1, false, false);
    __syncthreads();
  }

  // ---- epilogue: C/D layout: elem e -> M = e + 8*(lane>=16), N = lane&15 ----
  const int moff = (lane >> 4) * 8;
  const int mrow = mblk + mloc + moff;
  const int col0 = nblk + nloc + (lane & 15);
  const int col1 = col0 + 16;
  if (col0 < N) {
    float bv = bias ? bias[col0] : 0.0f;
    #pragma unroll
    for (int e = 0; e < 8; ++e) {
      int row = mrow + e;
      if (row < M) C[(long)row * N + col0] = fast_act(acc0[e] + bv, act);
    }
  }
  if (col1 < N) {
    float bv = bias ? bias[col1] : 0.0f;
    #pragma unroll
    for (int e = 0; e < 8; ++e) {
      int row = mrow + e;
      if (row < M) C[(long)row * N + col1] = fast_act(acc1[e] + bv, act);
    }
  }
}

// ---------------------------------------------------------------------------
// BatchNorm (training mode): per-channel mean/var over all M rows of [M,C]
// ---------------------------------------------------------------------------
__global__ __launch_bounds__(256) void bn_stats_kernel(
    const float* __restrict__ X, int M, int C,
    float* __restrict__ mean, float* __restrict__ var)
{
  __shared__ float s1[256], s2[256];
  int c = blockIdx.x, t = threadIdx.x;
  float a = 0.f, b = 0.f;
  for (int r = t; r < M; r += 256) {
    float v = X[(long)r * C + c];
    a += v; b += v * v;
  }
  s1[t] = a; s2[t] = b;
  __syncthreads();
  for (int s = 128; s > 0; s >>= 1) {
    if (t < s) { s1[t] += s1[t + s]; s2[t] += s2[t + s]; }
    __syncthreads();
  }
  if (t == 0) {
    float m = s1[0] / (float)M;
    mean[c] = m;
    var[c]  = s2[0] / (float)M - m * m;
  }
}

// y = g*(x-m)*rsqrt(v+eps)+b, then: skip? relu(skip+y) : act(y)
__global__ void bn_apply_kernel(
    const float* __restrict__ X, const float* __restrict__ mean,
    const float* __restrict__ var, const float* __restrict__ g,
    const float* __restrict__ b, const float* __restrict__ skip,
    float* __restrict__ Y, long tot, int C, int act)
{
  long i = (long)blockIdx.x * blockDim.x + threadIdx.x;
  if (i >= tot) return;
  int c = (int)(i % C);
  float y = g[c] * (X[i] - mean[c]) * rsqrtf(var[c] + 1e-5f) + b[c];
  if (skip) y = fmaxf(skip[i] + y, 0.0f);
  else      y = fast_act(y, act);
  Y[i] = y;
}

// [NB, PP, C] -> NCHW flat [NB, C, PP]
__global__ void npc_to_nchw_kernel(const float* __restrict__ src,
                                   float* __restrict__ dst, int C)
{
  long i = (long)blockIdx.x * blockDim.x + threadIdx.x;
  long tot = (long)NB * C * PP;
  if (i >= tot) return;
  int b = (int)(i / ((long)C * PP));
  int rem = (int)(i - (long)b * C * PP);
  int c = rem / PP, p = rem - c * PP;
  dst[i] = src[((long)(b * PP + p)) * C + c];
}

// OIHW [Cout,Cin,3,3] -> K-major [taps*Cin, Cout] with k = tap*Cin + c
__global__ void wreorder_kernel(const float* __restrict__ W,
                                float* __restrict__ Wt,
                                int Cout, int Cin, int taps)
{
  long i = (long)blockIdx.x * blockDim.x + threadIdx.x;
  long tot = (long)Cout * Cin * taps;
  if (i >= tot) return;
  int o = (int)(i % Cout);
  long k = i / Cout;
  int tap = (int)(k / Cin), c = (int)(k - (long)tap * Cin);
  Wt[i] = W[((long)o * Cin + c) * taps + tap];
}

// per-row squared norm of E[R,d]
__global__ void rownorm_kernel(const float* __restrict__ E,
                               float* __restrict__ n, int R, int d)
{
  int r = blockIdx.x * blockDim.x + threadIdx.x;
  if (r >= R) return;
  float s = 0.f;
  for (int t = 0; t < d; ++t) { float v = E[(long)r * d + t]; s += v * v; }
  n[r] = s;
}

// Fused VQ: one wave per row. dist = ||E||^2 - 2 lp.E  (|lp|^2 drops in argmin)
__global__ __launch_bounds__(256) void vq_kernel(
    const float* __restrict__ lp, const float* __restrict__ E,
    const float* __restrict__ En, int M, int d, int Kc,
    float* __restrict__ kout, float* __restrict__ zq, float* __restrict__ zql)
{
  __shared__ float slp[8][64];
  int wave = threadIdx.x >> 5, lane = threadIdx.x & 31;
  int row = blockIdx.x * 8 + wave;
  if (row < M)
    for (int t = lane; t < d; t += 32) slp[wave][t] = lp[(long)row * d + t];
  __syncthreads();
  if (row >= M) return;

  float best = 3.4e38f; int bidx = 0;
  for (int e = lane; e < Kc; e += 32) {
    const float* Er = E + (long)e * d;
    float dot = 0.f;
    for (int t = 0; t < d; ++t) dot = fmaf(slp[wave][t], Er[t], dot);
    float dist = En[e] - 2.f * dot;
    if (dist < best) { best = dist; bidx = e; }
  }
  for (int off = 16; off > 0; off >>= 1) {
    float ob = __shfl_xor(best, off, 32);
    int   oi = __shfl_xor(bidx, off, 32);
    if (ob < best || (ob == best && oi < bidx)) { best = ob; bidx = oi; }
  }
  if (lane == 0) kout[row] = (float)bidx;
  const float* Er = E + (long)bidx * d;
  for (int t = lane; t < d; t += 32) {
    float v = Er[t];
    zq[(long)row * d + t]  = v;
    zql[(long)row * d + t] = v;  // straight-through: forward value == zq
  }
}

// column mean of E[R,d] (d <= 64)
__global__ void colmean_kernel(const float* __restrict__ E,
                               float* __restrict__ m, int R, int d)
{
  int t = threadIdx.x;
  if (t >= d) return;
  float s = 0.f;
  for (int r = 0; r < R; ++r) s += E[(long)r * d + t];
  m[t] = s / (float)R;
}

// broadcast per-channel value: dst[i] = m[(i/inner) % C]
__global__ void bcast_em_kernel(const float* __restrict__ m,
                                float* __restrict__ dst,
                                long tot, int C, int inner)
{
  long i = (long)blockIdx.x * blockDim.x + threadIdx.x;
  if (i >= tot) return;
  dst[i] = m[(int)((i / inner) % C)];
}

// out = u * zql + (1-u) * x,  u = usage[b*2 + col], b = i / perB
__global__ void blend_kernel(const float* __restrict__ zql,
                             const float* __restrict__ x,
                             const float* __restrict__ us,
                             int col, int perB,
                             float* __restrict__ out, long tot)
{
  long i = (long)blockIdx.x * blockDim.x + threadIdx.x;
  if (i >= tot) return;
  int b = (int)(i / perB);
  float u = us[b * 2 + col];
  out[i] = u * zql[i] + (1.f - u) * x[i];
}

inline dim3 gemm_grid(int M, int N) { return dim3((N + 63) / 64, (M + 63) / 64); }
inline long cdiv(long a, long b) { return (a + b - 1) / b; }

} // namespace

extern "C" void kernel_launch(void* const* d_in, const int* in_sizes, int n_in,
                              void* d_out, int out_size, void* d_ws, size_t ws_size,
                              hipStream_t stream) {
  (void)in_sizes; (void)n_in; (void)out_size; (void)ws_size;
  const float* obs    = (const float*)d_in[0];
  const float* husg   = (const float*)d_in[1];
  const float* h0w    = (const float*)d_in[2];
  const float* h0b    = (const float*)d_in[3];
  const float* h0g    = (const float*)d_in[4];
  const float* h0bb   = (const float*)d_in[5];
  const float* rc1w   = (const float*)d_in[6];
  const float* rc1b   = (const float*)d_in[7];
  const float* rbn1g  = (const float*)d_in[8];
  const float* rbn1b  = (const float*)d_in[9];
  const float* rc2w   = (const float*)d_in[10];
  const float* rc2b   = (const float*)d_in[11];
  const float* rbn2g  = (const float*)d_in[12];
  const float* rbn2b  = (const float*)d_in[13];
  const float* h02w   = (const float*)d_in[14];
  const float* h02b   = (const float*)d_in[15];
  const float* h02g   = (const float*)d_in[16];
  const float* h02bb  = (const float*)d_in[17];
  const float* E0     = (const float*)d_in[18];
  const float* E1     = (const float*)d_in[19];
  const float* E2     = (const float*)d_in[20];
  const float* h1fcw  = (const float*)d_in[21];
  const float* h1fcb  = (const float*)d_in[22];
  const float* h1fc2w = (const float*)d_in[23];
  const float* h1fc2b = (const float*)d_in[24];
  const float* h2fcw  = (const float*)d_in[25];
  const float* h2fcb  = (const float*)d_in[26];
  const float* h2fc2w = (const float*)d_in[27];
  const float* h2fc2b = (const float*)d_in[28];
  const float* h2fc3w = (const float*)d_in[29];
  const float* h2fc3b = (const float*)d_in[30];
  const float* h2fc4w = (const float*)d_in[31];
  const float* h2fc4b = (const float*)d_in[32];
  const float* h1fc3w = (const float*)d_in[33];
  const float* h1fc3b = (const float*)d_in[34];
  const float* h1fc4w = (const float*)d_in[35];
  const float* h1fc4b = (const float*)d_in[36];
  const float* polw   = (const float*)d_in[37];
  const float* polb   = (const float*)d_in[38];
  const float* valw   = (const float*)d_in[39];
  const float* valb   = (const float*)d_in[40];

  // ---- d_out layout (tuple flat, return order) ----
  float* out = (float*)d_out;
  long off = 0;
  const long N_Z0 = (long)NB * 32 * PP;  // 1478656
  const long N_Z1 = (long)NB * 64 * 8;   // 65536
  float* o_policy = out + off; off += (long)NB * 362;
  float* o_value  = out + off; off += NB;
  float* o_ze0 = out + off; off += N_Z0;
  float* o_ze1 = out + off; off += N_Z1;
  float* o_ze2 = out + off; off += N_Z1;
  float* o_zq0 = out + off; off += N_Z0;
  float* o_zq1 = out + off; off += N_Z1;
  float* o_zq2 = out + off; off += N_Z1;
  float* o_k0  = out + off; off += (long)NB * PP;
  float* o_k1  = out + off; off += (long)NB * 8;
  float* o_k2  = out + off; off += (long)NB * 8;
  float* o_zql0 = out + off; off += N_Z0;
  float* o_zql1 = out + off; off += N_Z1;
  float* o_zql2 = out + off; off += N_Z1;
  float* o_em0 = out + off; off += N_Z0;
  float* o_em1 = out + off; off += N_Z1;
  float* o_em2 = out + off; off += N_Z1;

  // ---- workspace (bump allocator) ----
  float* ws = (float*)d_ws;
  long woff = 0;
  auto alloc = [&](long n) { float* p = ws + woff; woff += n; return p; };
  float* w_wt    = alloc((long)2304 * 256);
  float* w_raw   = alloc((long)MPIX * 256);
  float* w_actA  = alloc((long)MPIX * 256);
  float* w_actB  = alloc((long)MPIX * 256);
  float* w_z32   = alloc((long)MPIX * 32);
  float* w_mean  = alloc(256);
  float* w_var   = alloc(256);
  float* w_en0   = alloc(512);
  float* w_en1   = alloc(512);
  float* w_en2   = alloc(512);
  float* w_em    = alloc(64);
  float* w_f256  = alloc((long)NB * 256);
  float* w_f128  = alloc((long)NB * 128);
  float* w_d512  = alloc((long)NB * 512);
  float* w_x512  = alloc((long)NB * 512);
  float* w_f256b = alloc((long)NB * 256);
  float* w_xdec  = alloc((long)NB * 11552);
  float* w_xb0   = alloc((long)NB * 11552);

  auto GEMM = [&](const float* A, const float* Bw, const float* bias, float* Cst,
                  int M, int N, int K, int mode, int Cin, int act) {
    gemm_wmma_kernel<<<gemm_grid(M, N), 256, 0, stream>>>(A, Bw, bias, Cst,
                                                          M, N, K, mode, Cin, act);
  };
  auto BN = [&](const float* X, const float* g, const float* b,
                const float* skip, float* Y, int C, int act) {
    bn_stats_kernel<<<C, 256, 0, stream>>>(X, MPIX, C, w_mean, w_var);
    long tot = (long)MPIX * C;
    bn_apply_kernel<<<(int)cdiv(tot, 256), 256, 0, stream>>>(
        X, w_mean, w_var, g, b, skip, Y, tot, C, act);
  };

  // ---- conv stem: relu(bn(conv(obs))) ----
  wreorder_kernel<<<(int)cdiv((long)162 * 256, 256), 256, 0, stream>>>(h0w, w_wt, 256, 18, 9);
  GEMM(obs, w_wt, h0b, w_raw, MPIX, 256, 162, /*mode NCHW*/2, 18, 0);
  BN(w_raw, h0g, h0bb, nullptr, w_actA, 256, 1);

  // ---- 3 residual blocks ----
  for (int i = 0; i < 3; ++i) {
    const long wsz = (long)256 * 256 * 9;
    wreorder_kernel<<<(int)cdiv(wsz, 256), 256, 0, stream>>>(rc1w + i * wsz, w_wt, 256, 256, 9);
    GEMM(w_actA, w_wt, rc1b + i * 256, w_raw, MPIX, 256, 2304, 1, 256, 0);
    BN(w_raw, rbn1g + i * 256, rbn1b + i * 256, nullptr, w_actB, 256, 1);
    wreorder_kernel<<<(int)cdiv(wsz, 256), 256, 0, stream>>>(rc2w + i * wsz, w_wt, 256, 256, 9);
    GEMM(w_actB, w_wt, rc2b + i * 256, w_raw, MPIX, 256, 2304, 1, 256, 0);
    BN(w_raw, rbn2g + i * 256, rbn2b + i * 256, /*skip*/w_actA, w_actA, 256, 0);
  }

  // ---- tanh(bn(conv2)) -> z [B,32,19,19] ----
  wreorder_kernel<<<(int)cdiv((long)2304 * 32, 256), 256, 0, stream>>>(h02w, w_wt, 32, 256, 9);
  GEMM(w_actA, w_wt, h02b, w_raw, MPIX, 32, 2304, 1, 256, 0);
  BN(w_raw, h02g, h02bb, nullptr, w_z32, 32, 2);
  npc_to_nchw_kernel<<<(int)cdiv(N_Z0, 256), 256, 0, stream>>>(w_z32, o_ze0, 32);

  // ---- codebook norms + VQ0 ----
  rownorm_kernel<<<2, 256, 0, stream>>>(E0, w_en0, 512, 32);
  rownorm_kernel<<<2, 256, 0, stream>>>(E1, w_en1, 512, 64);
  rownorm_kernel<<<2, 256, 0, stream>>>(E2, w_en2, 512, 64);
  vq_kernel<<<(MPIX + 7) / 8, 256, 0, stream>>>(o_ze0, E0, w_en0, MPIX, 32, 512,
                                                o_k0, o_zq0, o_zql0);

  // ---- embedding means broadcast ----
  colmean_kernel<<<1, 64, 0, stream>>>(E0, w_em, 512, 32);
  bcast_em_kernel<<<(int)cdiv(N_Z0, 256), 256, 0, stream>>>(w_em, o_em0, N_Z0, 32, PP);
  colmean_kernel<<<1, 64, 0, stream>>>(E1, w_em, 512, 64);
  bcast_em_kernel<<<(int)cdiv(N_Z1, 256), 256, 0, stream>>>(w_em, o_em1, N_Z1, 64, 8);
  colmean_kernel<<<1, 64, 0, stream>>>(E2, w_em, 512, 64);
  bcast_em_kernel<<<(int)cdiv(N_Z1, 256), 256, 0, stream>>>(w_em, o_em2, N_Z1, 64, 8);

  // ---- encoder FC ladder ----
  GEMM(o_ze0, h1fcw, h1fcb, w_f256, NB, 256, 11552, 0, 0, 1);
  GEMM(w_f256, h1fc2w, h1fc2b, o_ze1, NB, 512, 256, 0, 0, 2);
  vq_kernel<<<128, 256, 0, stream>>>(o_ze1, E1, w_en1, 1024, 64, 512, o_k1, o_zq1, o_zql1);
  GEMM(o_ze1, h2fcw, h2fcb, w_f128, NB, 128, 512, 0, 0, 1);
  GEMM(w_f128, h2fc2w, h2fc2b, o_ze2, NB, 512, 128, 0, 0, 2);
  vq_kernel<<<128, 256, 0, stream>>>(o_ze2, E2, w_en2, 1024, 64, 512, o_k2, o_zq2, o_zql2);

  // ---- top-down decoder ----
  GEMM(o_zql2, h2fc3w, h2fc3b, w_f128, NB, 128, 512, 0, 0, 1);
  GEMM(w_f128, h2fc4w, h2fc4b, w_d512, NB, 512, 128, 0, 0, 2);
  blend_kernel<<<(int)cdiv(N_Z1, 256), 256, 0, stream>>>(o_zql1, w_d512, husg, 1, 512,
                                                         w_x512, N_Z1);
  GEMM(w_x512, h1fc3w, h1fc3b, w_f256b, NB, 256, 512, 0, 0, 1);
  GEMM(w_f256b, h1fc4w, h1fc4b, w_xdec, NB, 11552, 256, 0, 0, 2);
  blend_kernel<<<(int)cdiv(N_Z0, 256), 256, 0, stream>>>(o_zql0, w_xdec, husg, 0, 11552,
                                                         w_xb0, N_Z0);

  // ---- heads ----
  GEMM(w_xb0, polw, polb, o_policy, NB, 362, 11552, 0, 0, 0);
  GEMM(w_xb0, valw, valb, o_value, NB, 1, 11552, 0, 0, 2);
}
